// Transformer_71760313582240
// MI455X (gfx1250) — matrix operand order repaired and checked by reading, not compiled, not run on hardware
//
#include <hip/hip_runtime.h>
#include <hip/hip_bf16.h>
#include <math.h>

// ---------------------------------------------------------------------------
// Transformer block for MI455X (gfx1250): bf16 WMMA GEMMs (async-LDS double
// buffered) + flash attention.
// ---------------------------------------------------------------------------

typedef __bf16 v16bf __attribute__((ext_vector_type(16)));
typedef __bf16 v8bf  __attribute__((ext_vector_type(8)));
typedef float  v8f   __attribute__((ext_vector_type(8)));

union ABf { v16bf v; v8bf h[2]; };

#define EMB   1024
#define HIDN  4096
#define HEADS 16
#define HD    64
#define SEQ   2048
#define TOK   4096   // B*S = 2*2048

static __device__ __forceinline__ v8f wmma_bf16(const v16bf& a, const v16bf& b, v8f c) {
    return __builtin_amdgcn_wmma_f32_16x16x32_bf16(false, a, false, b, (short)0, c,
                                                   false, false);
}

// LDS byte offset of a generic pointer to a __shared__ object: flat LDS-aperture
// addresses carry the LDS offset in bits [31:0].
static __device__ __forceinline__ unsigned lds_off(const void* p) {
    return (unsigned)(unsigned long long)p;
}

// CDNA5 async copy global -> LDS (16 bytes per lane), tracked by ASYNCcnt.
static __device__ __forceinline__ void async_ld_b128(unsigned lds, const void* g) {
    asm volatile("global_load_async_to_lds_b128 %0, %1, off"
                 :: "v"(lds), "v"(g) : "memory");
}
static __device__ __forceinline__ void wait_async0() {
    asm volatile("s_wait_asynccnt 0x0" ::: "memory");
}

// ------------------------- weight pack: fp32 KxN -> bf16 NxK ----------------
__global__ void pack_wT_kernel(const float* __restrict__ W, __bf16* __restrict__ WT,
                               int K, int N) {
    int i = blockIdx.x * 256 + threadIdx.x;
    int total = K * N;
    if (i < total) {
        int n = i / K;
        int k = i - n * K;
        WT[i] = (__bf16)W[(size_t)k * N + n];
    }
}

// ------------------------- layernorm -> bf16 --------------------------------
__global__ __launch_bounds__(256)
void ln_kernel(const float* __restrict__ x, const float* __restrict__ scale,
               const float* __restrict__ shift, __bf16* __restrict__ out) {
    int row = blockIdx.x;
    int tid = threadIdx.x;
    const float* xr = x + (size_t)row * EMB;
    float4 v4 = reinterpret_cast<const float4*>(xr)[tid];
    float vv[4] = {v4.x, v4.y, v4.z, v4.w};
    float s = vv[0] + vv[1] + vv[2] + vv[3];
    float s2 = vv[0]*vv[0] + vv[1]*vv[1] + vv[2]*vv[2] + vv[3]*vv[3];

    __shared__ float redS[256];
    __shared__ float redQ[256];
    redS[tid] = s; redQ[tid] = s2;
    __syncthreads();
    for (int off = 128; off > 0; off >>= 1) {
        if (tid < off) { redS[tid] += redS[tid + off]; redQ[tid] += redQ[tid + off]; }
        __syncthreads();
    }
    float mean = redS[0] * (1.0f / EMB);
    float var  = redQ[0] * (1.0f / EMB) - mean * mean;
    float inv  = rsqrtf(var + 1e-5f);

    __bf16* orow = out + (size_t)row * EMB;
    for (int j = 0; j < 4; j++) {
        int c = tid * 4 + j;
        orow[c] = (__bf16)((vv[j] - mean) * inv * scale[c] + shift[c]);
    }
}

// ------------------------- generic bf16 WMMA GEMM ---------------------------
// C[M,N] = A[M,K] (bf16 row-major) * B, with B given as BT[N,K] (bf16 row-major).
// Double-buffered LDS tiles filled with async global->LDS copies.
// mode 0: out bf16 plain   [m*N+n]
// mode 1: out bf16 Q/K head-split  [((b*16+h)*2048+s)*64+d]
// mode 2: out bf16 V transposed    [((b*16+h)*64+d)*2048+s]
// mode 3: out fp32 = resid + C + bias
// mode 4: out bf16 = gelu(C + bias)
__global__ __launch_bounds__(256)
void gemm_kernel(const __bf16* __restrict__ A, const __bf16* __restrict__ BT,
                 const float* __restrict__ bias, const float* __restrict__ resid,
                 float* __restrict__ outF, __bf16* __restrict__ outB,
                 int M, int N, int K, int mode) {
    __shared__ __align__(16) __bf16 As[2][128 * 40];
    __shared__ __align__(16) __bf16 Bs[2][128 * 40];

    int tid  = threadIdx.x;
    int lane = tid & 31;
    int wid  = tid >> 5;
    int hf   = lane >> 4;       // half of wave
    int l16  = lane & 15;
    int wm   = wid >> 1;        // 0..3 (wave row)
    int wn   = wid & 1;         // 0..1 (wave col)
    int m0   = blockIdx.y * 128;
    int n0   = blockIdx.x * 128;

    // staging coordinates: two 16-byte chunks per thread per tile
    int chunk0 = tid * 2;              // 0..510
    int row0   = chunk0 >> 2;          // 0..127
    int cc0    = chunk0 & 3;           // 0..3
    int row1   = (chunk0 + 1) >> 2;
    int cc1    = (chunk0 + 1) & 3;

    const __bf16* Ag0 = A  + (size_t)(m0 + row0) * K + cc0 * 8;
    const __bf16* Ag1 = A  + (size_t)(m0 + row1) * K + cc1 * 8;
    const __bf16* Bg0 = BT + (size_t)(n0 + row0) * K + cc0 * 8;
    const __bf16* Bg1 = BT + (size_t)(n0 + row1) * K + cc1 * 8;

    unsigned la0[2], la1[2], lb0[2], lb1[2];
    for (int b = 0; b < 2; b++) {
        la0[b] = lds_off(&As[b][row0 * 40 + cc0 * 8]);
        la1[b] = lds_off(&As[b][row1 * 40 + cc1 * 8]);
        lb0[b] = lds_off(&Bs[b][row0 * 40 + cc0 * 8]);
        lb1[b] = lds_off(&Bs[b][row1 * 40 + cc1 * 8]);
    }

    v8f acc[2][4];
    for (int mi = 0; mi < 2; mi++)
        for (int ni = 0; ni < 4; ni++)
            acc[mi][ni] = (v8f){0,0,0,0,0,0,0,0};

    // prologue: stage first K-tile into buffer 0
    async_ld_b128(la0[0], Ag0);
    async_ld_b128(la1[0], Ag1);
    async_ld_b128(lb0[0], Bg0);
    async_ld_b128(lb1[0], Bg1);
    wait_async0();
    __syncthreads();

    for (int k0 = 0; k0 < K; k0 += 32) {
        int buf = (k0 >> 5) & 1;
        // prefetch next K-tile into the other buffer (async, no VGPR staging)
        if (k0 + 32 < K) {
            int nb = buf ^ 1;
            int kn = k0 + 32;
            async_ld_b128(la0[nb], Ag0 + kn);
            async_ld_b128(la1[nb], Ag1 + kn);
            async_ld_b128(lb0[nb], Bg0 + kn);
            async_ld_b128(lb1[nb], Bg1 + kn);
        }

        ABf af[2], bfv[4];
        for (int mi = 0; mi < 2; mi++) {
            int r = wm * 32 + mi * 16 + l16;
            af[mi].h[0] = *(const v8bf*)&As[buf][r * 40 + hf * 8];        // K 0-7 / 8-15
            af[mi].h[1] = *(const v8bf*)&As[buf][r * 40 + 16 + hf * 8];   // K 16-23 / 24-31
        }
        for (int ni = 0; ni < 4; ni++) {
            int r = wn * 64 + ni * 16 + l16;
            bfv[ni].h[0] = *(const v8bf*)&Bs[buf][r * 40 + hf * 8];       // K 0-7 / 8-15
            bfv[ni].h[1] = *(const v8bf*)&Bs[buf][r * 40 + 16 + hf * 8];
        }
        for (int mi = 0; mi < 2; mi++)
            for (int ni = 0; ni < 4; ni++)
                acc[mi][ni] = wmma_bf16(af[mi].v, bfv[ni].v, acc[mi][ni]);

        wait_async0();     // own async copies have landed in LDS
        __syncthreads();   // everyone's copies visible; reads of this buf done
    }

    // epilogue
    for (int mi = 0; mi < 2; mi++) {
        for (int ni = 0; ni < 4; ni++) {
            for (int r = 0; r < 8; r++) {
                int gm = m0 + wm * 32 + mi * 16 + 8 * hf + r;
                int gn = n0 + wn * 64 + ni * 16 + l16;
                float v = acc[mi][ni][r];
                if (mode == 0) {
                    outB[(size_t)gm * N + gn] = (__bf16)v;
                } else if (mode == 1) {
                    int b = gm >> 11, s = gm & 2047, hh = gn >> 6, d = gn & 63;
                    outB[(((size_t)(b * HEADS + hh) * SEQ) + s) * HD + d] = (__bf16)v;
                } else if (mode == 2) {
                    int b = gm >> 11, s = gm & 2047, hh = gn >> 6, d = gn & 63;
                    outB[(((size_t)(b * HEADS + hh) * HD) + d) * SEQ + s] = (__bf16)v;
                } else if (mode == 3) {
                    size_t idx = (size_t)gm * N + gn;
                    outF[idx] = resid[idx] + v + bias[gn];
                } else { // mode 4: gelu
                    float xg = v + bias[gn];
                    float t = tanhf(0.7978845608028654f * (xg + 0.044715f * xg * xg * xg));
                    outB[(size_t)gm * N + gn] = (__bf16)(0.5f * xg * (1.0f + t));
                }
            }
        }
    }
}

// ------------------------- flash attention ----------------------------------
// q,k : bf16 [b][h][s][64]   vT : bf16 [b][h][64][s]   out: bf16 [tok][1024]
__global__ __launch_bounds__(128)
void attn_kernel(const __bf16* __restrict__ q, const __bf16* __restrict__ k,
                 const __bf16* __restrict__ vT, __bf16* __restrict__ ctxm) {
    int bh   = blockIdx.y;                 // 0..31  (b*16+h)
    int w    = threadIdx.x >> 5;           // wave 0..3
    int lane = threadIdx.x & 31;
    int hf   = lane >> 4;
    int l16  = lane & 15;
    int q0   = (blockIdx.x * 4 + w) * 16;  // query tile start

    const __bf16* qb = q  + (size_t)bh * SEQ * HD;
    const __bf16* kb = k  + (size_t)bh * SEQ * HD;
    const __bf16* vb = vT + (size_t)bh * HD * SEQ;

    __shared__ __align__(16) __bf16 p_lds[4][16 * 32];
    __bf16* pl = p_lds[w];

    // Q fragments (A layout), constant over key loop
    ABf qa[2];
    {
        const __bf16* qrow = qb + (size_t)(q0 + l16) * HD;
        for (int c = 0; c < 2; c++) {
            int kc = c * 32;
            qa[c].h[0] = *(const v8bf*)(qrow + kc + hf * 8);
            qa[c].h[1] = *(const v8bf*)(qrow + kc + 16 + hf * 8);
        }
    }

    v8f ctx[4];
    for (int t = 0; t < 4; t++) ctx[t] = (v8f){0,0,0,0,0,0,0,0};
    float mi[8], li[8];
    for (int r = 0; r < 8; r++) { mi[r] = -1e30f; li[r] = 0.0f; }

    int nkb = (q0 + 16 + 31) >> 5;
    for (int ib = 0; ib < nkb; ib++) {
        int kb0 = ib * 32;

        // scores: two 16x16 key tiles, K=64 head dim over 2 k-steps
        v8f sc[2];
        sc[0] = (v8f){0,0,0,0,0,0,0,0};
        sc[1] = (v8f){0,0,0,0,0,0,0,0};
        for (int t = 0; t < 2; t++) {
            int n0t = kb0 + t * 16;
            for (int c = 0; c < 2; c++) {
                int kc = c * 32;
                ABf bfK;
                const __bf16* krow = kb + (size_t)(n0t + l16) * HD + kc + hf * 8;
                bfK.h[0] = *(const v8bf*)(krow);
                bfK.h[1] = *(const v8bf*)(krow + 16);
                sc[t] = wmma_bf16(qa[c].v, bfK.v, sc[t]);
            }
        }

        bool need_mask = (kb0 + 31) > q0;
        int key0 = kb0 + l16;
        int key1 = kb0 + 16 + l16;

        // online softmax per row (row m = 8*hf + r lives in one 16-lane half)
        for (int r = 0; r < 8; r++) {
            float a0 = sc[0][r] * 0.125f;
            float a1 = sc[1][r] * 0.125f;
            if (need_mask) {
                int qrow = q0 + 8 * hf + r;
                if (key0 > qrow) a0 = -1e30f;
                if (key1 > qrow) a1 = -1e30f;
            }
            float mx = fmaxf(a0, a1);
            for (int d = 1; d < 16; d <<= 1) mx = fmaxf(mx, __shfl_xor(mx, d, 32));
            float mnew  = fmaxf(mi[r], mx);
            float alpha = __expf(mi[r] - mnew);
            float p0 = __expf(a0 - mnew);
            float p1 = __expf(a1 - mnew);
            float ps = p0 + p1;
            for (int d = 1; d < 16; d <<= 1) ps += __shfl_xor(ps, d, 32);
            li[r] = li[r] * alpha + ps;
            mi[r] = mnew;
            for (int t = 0; t < 4; t++) ctx[t][r] *= alpha;
            int m = 8 * hf + r;
            pl[m * 32 + l16]      = (__bf16)p0;
            pl[m * 32 + 16 + l16] = (__bf16)p1;
        }
        asm volatile("s_wait_dscnt 0" ::: "memory");

        // re-read P in A-fragment layout (cross-lane transpose via LDS)
        ABf pa;
        pa.h[0] = *(const v8bf*)(pl + l16 * 32 + hf * 8);
        pa.h[1] = *(const v8bf*)(pl + l16 * 32 + 16 + hf * 8);

        // ctx += P (16x32) x V (32x64)
        for (int t = 0; t < 4; t++) {
            int d0 = t * 16;
            ABf bfV;
            const __bf16* vrow = vb + (size_t)(d0 + l16) * SEQ + kb0 + hf * 8;
            bfV.h[0] = *(const v8bf*)(vrow);
            bfV.h[1] = *(const v8bf*)(vrow + 16);
            ctx[t] = wmma_bf16(pa.v, bfV.v, ctx[t]);
        }
        asm volatile("s_wait_dscnt 0" ::: "memory");
    }

    // normalize + store merged [tok][1024] bf16
    int b = bh >> 4, hh = bh & 15;
    for (int r = 0; r < 8; r++) {
        float invl = 1.0f / li[r];
        int m = q0 + 8 * hf + r;
        size_t base = ((size_t)(b * SEQ + m)) * EMB + hh * HD;
        for (int t = 0; t < 4; t++) {
            ctxm[base + t * 16 + l16] = (__bf16)(ctx[t][r] * invl);
        }
    }
}

// ------------------------- host-side orchestration --------------------------
#define MB (1024ull * 1024ull)

extern "C" void kernel_launch(void* const* d_in, const int* in_sizes, int n_in,
                              void* d_out, int out_size, void* d_ws, size_t ws_size,
                              hipStream_t stream) {
    const float* x   = (const float*)d_in[0];
    const float* Wq  = (const float*)d_in[1];
    const float* Wk  = (const float*)d_in[2];
    const float* Wv  = (const float*)d_in[3];
    const float* Wo  = (const float*)d_in[4];
    const float* bo  = (const float*)d_in[5];
    const float* W1  = (const float*)d_in[6];
    const float* b1  = (const float*)d_in[7];
    const float* W2  = (const float*)d_in[8];
    const float* b2  = (const float*)d_in[9];
    const float* l1s = (const float*)d_in[10];
    const float* l1h = (const float*)d_in[11];
    const float* l2s = (const float*)d_in[12];
    const float* l2h = (const float*)d_in[13];
    float* out = (float*)d_out;

    char* ws = (char*)d_ws;
    __bf16* h_bf = (__bf16*)(ws + 0);          //  8 MB  (reused for ln2 output)
    __bf16* WqT  = (__bf16*)(ws + 8  * MB);    //  2 MB
    __bf16* WkT  = (__bf16*)(ws + 10 * MB);    //  2 MB
    __bf16* WvT  = (__bf16*)(ws + 12 * MB);    //  2 MB
    __bf16* WoT  = (__bf16*)(ws + 14 * MB);    //  2 MB
    __bf16* W1T  = (__bf16*)(ws + 16 * MB);    //  8 MB
    __bf16* W2T  = (__bf16*)(ws + 24 * MB);    //  8 MB
    __bf16* qb   = (__bf16*)(ws + 32 * MB);    //  8 MB
    __bf16* kbuf = (__bf16*)(ws + 40 * MB);    //  8 MB
    __bf16* vTb  = (__bf16*)(ws + 48 * MB);    //  8 MB
    __bf16* ctxm = (__bf16*)(ws + 56 * MB);    //  8 MB
    float*  x1   = (float*) (ws + 64 * MB);    // 16 MB
    __bf16* hid  = (__bf16*)(ws + 32 * MB);    // 32 MB (aliases q/k/vT/ctx after O-proj)

    // 1) pack weights -> bf16 transposed (NxK)
    int nEE = EMB * EMB;
    pack_wT_kernel<<<(nEE + 255) / 256, 256, 0, stream>>>(Wq, WqT, EMB, EMB);
    pack_wT_kernel<<<(nEE + 255) / 256, 256, 0, stream>>>(Wk, WkT, EMB, EMB);
    pack_wT_kernel<<<(nEE + 255) / 256, 256, 0, stream>>>(Wv, WvT, EMB, EMB);
    pack_wT_kernel<<<(nEE + 255) / 256, 256, 0, stream>>>(Wo, WoT, EMB, EMB);
    int nEH = EMB * HIDN;
    pack_wT_kernel<<<(nEH + 255) / 256, 256, 0, stream>>>(W1, W1T, EMB, HIDN);
    pack_wT_kernel<<<(nEH + 255) / 256, 256, 0, stream>>>(W2, W2T, HIDN, EMB);

    // 2) ln1 -> h (bf16)
    ln_kernel<<<TOK, 256, 0, stream>>>(x, l1s, l1h, h_bf);

    // 3) Q/K/V projections (head-split stores; V transposed)
    dim3 gE(EMB / 128, TOK / 128);   // (8, 32)
    gemm_kernel<<<gE, 256, 0, stream>>>(h_bf, WqT, nullptr, nullptr, nullptr, qb,
                                        TOK, EMB, EMB, 1);
    gemm_kernel<<<gE, 256, 0, stream>>>(h_bf, WkT, nullptr, nullptr, nullptr, kbuf,
                                        TOK, EMB, EMB, 1);
    gemm_kernel<<<gE, 256, 0, stream>>>(h_bf, WvT, nullptr, nullptr, nullptr, vTb,
                                        TOK, EMB, EMB, 2);

    // 4) fused causal attention -> merged ctx (bf16)
    attn_kernel<<<dim3(SEQ / 16 / 4, 2 * HEADS), 128, 0, stream>>>(qb, kbuf, vTb, ctxm);

    // 5) O projection + residual: x1 = x + ctx@Wo + bo  (fp32)
    gemm_kernel<<<gE, 256, 0, stream>>>(ctxm, WoT, bo, x, x1, nullptr,
                                        TOK, EMB, EMB, 3);

    // 6) ln2 -> h (bf16, reuse)
    ln_kernel<<<TOK, 256, 0, stream>>>(x1, l2s, l2h, h_bf);

    // 7) FFN1 + gelu -> hid (bf16)
    dim3 gH(HIDN / 128, TOK / 128);  // (32, 32)
    gemm_kernel<<<gH, 256, 0, stream>>>(h_bf, W1T, b1, nullptr, nullptr, hid,
                                        TOK, HIDN, EMB, 4);

    // 8) FFN2 + residual -> out (fp32)
    gemm_kernel<<<gE, 256, 0, stream>>>(hid, W2T, b2, x1, out, nullptr,
                                        TOK, EMB, HIDN, 3);
}
